// FaRMamba_56075093016888
// MI455X (gfx1250) — matrix operand order repaired and checked by reading, not compiled, not run on hardware
//
#include <hip/hip_runtime.h>
#include <hip/hip_bf16.h>

typedef _Float16 half_t;
typedef __attribute__((ext_vector_type(16))) _Float16 v16h;
typedef __attribute__((ext_vector_type(8)))  _Float16 v8h;
typedef __attribute__((ext_vector_type(8)))  float    v8f;
typedef __attribute__((ext_vector_type(4)))  float    v4f;

#define WMMA_F32_F16(a, b, c) \
  __builtin_amdgcn_wmma_f32_16x16x32_f16(false, (a), false, (b), (short)0, (c), false, false)

static constexpr int C_DIM   = 256;
static constexpr int NH      = 8;
static constexpr int HD      = 32;
static constexpr int WS      = 8;
static constexpr int NTOK    = 64;
static constexpr int BATCH   = 8;
static constexpr int HGT     = 128;
static constexpr int WID     = 128;
static constexpr int NWIN    = BATCH * (HGT / WS) * (WID / WS);   // 2048
static constexpr int MROWS   = NWIN * NTOK;                        // 131072
static constexpr int HIDDEN  = 1024;

// ---------------------------------------------------------------------------
// f32 -> f16 conversion (weights)
// ---------------------------------------------------------------------------
__global__ void cvt_f16_kernel(const float* __restrict__ in, half_t* __restrict__ out, int n) {
  int i = blockIdx.x * 256 + threadIdx.x;
  if (i < n) out[i] = (half_t)in[i];
}

// ---------------------------------------------------------------------------
// Expand relative-position bias: out[head][n][m] = rel_table[rel_idx[n*64+m]][head]
// Removes the dependent 2-load chain from the attention softmax loop.
// ---------------------------------------------------------------------------
__global__ void bias_expand_kernel(const float* __restrict__ rel_table,
                                   const int* __restrict__ rel_idx,
                                   float* __restrict__ bias) {
  int i = blockIdx.x * 256 + threadIdx.x;            // 0..4095 = n*64+m
  if (i < NTOK * NTOK) {
    int id = rel_idx[i];
    #pragma unroll
    for (int h = 0; h < NH; ++h) bias[h * (NTOK * NTOK) + i] = rel_table[id * NH + h];
  }
}

// ---------------------------------------------------------------------------
// LayerNorm over C=256, one row per 256-thread block.
// windowed=1 also applies the Swin window-partition permutation.
// ---------------------------------------------------------------------------
__global__ __launch_bounds__(256) void ln_kernel(const float* __restrict__ x,
                                                 const float* __restrict__ g,
                                                 const float* __restrict__ b,
                                                 half_t* __restrict__ out,
                                                 int windowed) {
  __shared__ float red[256];
  const int r = blockIdx.x;
  const int c = threadIdx.x;

  int src;
  if (windowed) {
    int win = r >> 6, n = r & 63;
    int bb  = win >> 8, rem = win & 255;
    int wy  = rem >> 4, wx = rem & 15;
    int iy  = n >> 3,  ix = n & 7;
    src = (bb << 14) + (wy * WS + iy) * WID + (wx * WS + ix);
  } else {
    src = r;
  }

  const float v = x[(size_t)src * C_DIM + c];
  red[c] = v;
  __syncthreads();
  #pragma unroll
  for (int s = 128; s > 0; s >>= 1) { if (c < s) red[c] += red[c + s]; __syncthreads(); }
  const float mean = red[0] * (1.0f / 256.0f);
  __syncthreads();
  const float d = v - mean;
  red[c] = d * d;
  __syncthreads();
  #pragma unroll
  for (int s = 128; s > 0; s >>= 1) { if (c < s) red[c] += red[c + s]; __syncthreads(); }
  const float var = red[0] * (1.0f / 256.0f);

  out[(size_t)r * C_DIM + c] = (half_t)(d * rsqrtf(var + 1e-5f) * g[c] + b[c]);
}

// ---------------------------------------------------------------------------
// WMMA GEMM: C[M,N] = A[M,K](f16) x B[K,N](f16) + bias, fused epilogue.
//   Block tile 128(M) x 64(N), K step 32; 8 waves, each 32x32 (4 WMMAs/step).
//   B tile transposed in LDS -> contiguous b128 B-fragment loads.
//   Epilogue staged through LDS for fully coalesced b128 stores.
// MODE 0: QKV  -> f16 scattered per (win,head): contiguous [64][32] blocks
// MODE 1: PROJ -> f32 out = resid + val, window-reverse token mapping
// MODE 2: FC1  -> f16 out = ReLU6(val), row-major [M,1024]
// MODE 3: FC2  -> f32 out[idx] += val (residual accumulate)
// ---------------------------------------------------------------------------
template <int MODE>
__global__ __launch_bounds__(256) void gemm_wmma_kernel(const half_t* __restrict__ Ag,
                                                        const half_t* __restrict__ Bg,
                                                        const float*  __restrict__ bias,
                                                        void* __restrict__ outp,
                                                        const float* __restrict__ resid,
                                                        int M, int N, int K) {
  __shared__ __align__(16) unsigned char smem[16384];   // Al(8K)+Bt(5K) then C-stage(16K)
  half_t* Al = (half_t*)smem;                // [128][32]
  half_t* Bt = (half_t*)(smem + 8192);       // [64][40] transposed (16B-aligned rows)

  const int tid  = threadIdx.x;
  const int lane = tid & 31;
  const int wid  = tid >> 5;
  const int wm   = wid >> 1;                 // 0..3 : 32-row group
  const int wn   = wid & 1;                  // 0..1 : 32-col group
  const int rowBase = blockIdx.y * 128;
  const int colBase = blockIdx.x * 64;

  const int aRow = tid >> 1, aOff = (tid & 1) * 16;   // A loader: 2 x v8h per thread
  const int bRow = tid >> 3, bSeg = tid & 7;          // B loader

  v8f acc[2][2];
  { v8f z = {}; acc[0][0] = z; acc[0][1] = z; acc[1][0] = z; acc[1][1] = z; }

  const int ad0 = (lane < 16) ? 0 : 16;
  const int br0 = (lane < 16) ? 0 : 16;
  const int nLoc = lane & 15;

  for (int k0 = 0; k0 < K; k0 += 32) {
    *(v8h*)&Al[aRow * 32 + aOff]     = *(const v8h*)(Ag + (size_t)(rowBase + aRow) * K + k0 + aOff);
    *(v8h*)&Al[aRow * 32 + aOff + 8] = *(const v8h*)(Ag + (size_t)(rowBase + aRow) * K + k0 + aOff + 8);
    {
      v8h bv = *(const v8h*)(Bg + (size_t)(k0 + bRow) * N + colBase + bSeg * 8);
      #pragma unroll
      for (int i = 0; i < 8; ++i) Bt[(bSeg * 8 + i) * 40 + bRow] = bv[i];
    }
    if (k0 + 32 < K) {
      __builtin_prefetch(Ag + (size_t)(rowBase + aRow) * K + k0 + 32 + aOff, 0, 1);
      __builtin_prefetch(Bg + (size_t)(k0 + 32 + bRow) * N + colBase + bSeg * 8, 0, 1);
    }
    __syncthreads();

    v16h af[2], bf[2];
    #pragma unroll
    for (int tm = 0; tm < 2; ++tm) {
      const int arow = wm * 32 + tm * 16 + nLoc;
      v8h lo = *(const v8h*)&Al[arow * 32 + ad0];
      v8h hi = *(const v8h*)&Al[arow * 32 + ad0 + 8];
      #pragma unroll
      for (int i = 0; i < 8; ++i) { af[tm][i] = lo[i]; af[tm][i + 8] = hi[i]; }
    }
    #pragma unroll
    for (int tn = 0; tn < 2; ++tn)
      bf[tn] = *(const v16h*)&Bt[(wn * 32 + tn * 16 + nLoc) * 40 + br0];

    #pragma unroll
    for (int tm = 0; tm < 2; ++tm)
      #pragma unroll
      for (int tn = 0; tn < 2; ++tn)
        acc[tm][tn] = WMMA_F32_F16(af[tm], bf[tn], acc[tm][tn]);
    __syncthreads();
  }

  // ------------------- epilogue (LDS-staged, coalesced) -------------------
  const int mAdd = (lane < 16) ? 0 : 8;

  if constexpr (MODE == 0 || MODE == 2) {
    half_t* Cs = (half_t*)smem;                        // [128][64] f16 = 16KB
    #pragma unroll
    for (int tm = 0; tm < 2; ++tm)
      #pragma unroll
      for (int tn = 0; tn < 2; ++tn)
        #pragma unroll
        for (int v = 0; v < 8; ++v) {
          const int r = wm * 32 + tm * 16 + v + mAdd;
          const int c = wn * 32 + tn * 16 + nLoc;
          float val = acc[tm][tn][v] + bias[colBase + c];
          if constexpr (MODE == 2) val = fminf(fmaxf(val, 0.0f), 6.0f);
          Cs[r * 64 + c] = (half_t)val;
        }
    __syncthreads();
    #pragma unroll
    for (int j = 0; j < 4; ++j) {
      const int s  = tid * 4 + j;                      // 1024 v8h segments
      const int r  = s >> 3, c0 = (s & 7) * 8;
      v8h val = *(const v8h*)&Cs[r * 64 + c0];
      const int gr = rowBase + r, gc = colBase + c0;
      if constexpr (MODE == 0) {
        const int which = gc >> 8, head = (gc >> 5) & 7, d0 = gc & 31;
        const int win = gr >> 6, n = gr & 63;
        *(v8h*)((half_t*)outp +
                (((size_t)(win * 3 + which) * NH + head) * NTOK + n) * HD + d0) = val;
      } else {
        *(v8h*)((half_t*)outp + (size_t)gr * HIDDEN + gc) = val;
      }
    }
  } else {
    float* Cs = (float*)smem;                          // [64][64] f32 = 16KB, 2 passes
    #pragma unroll
    for (int pass = 0; pass < 2; ++pass) {
      if ((wm >> 1) == pass) {
        #pragma unroll
        for (int tm = 0; tm < 2; ++tm)
          #pragma unroll
          for (int tn = 0; tn < 2; ++tn)
            #pragma unroll
            for (int v = 0; v < 8; ++v) {
              const int rl = (wm & 1) * 32 + tm * 16 + v + mAdd;
              const int c  = wn * 32 + tn * 16 + nLoc;
              Cs[rl * 64 + c] = acc[tm][tn][v] + bias[colBase + c];
            }
      }
      __syncthreads();
      #pragma unroll
      for (int j = 0; j < 4; ++j) {
        const int s  = tid * 4 + j;                    // 1024 v4f segments
        const int r  = s >> 4, c0 = (s & 15) * 4;
        v4f val = *(const v4f*)&Cs[r * 64 + c0];
        const int gr = rowBase + pass * 64 + r, gc = colBase + c0;
        if constexpr (MODE == 1) {
          const int win = gr >> 6, n = gr & 63;
          const int bb  = win >> 8, rem = win & 255;
          const int wy  = rem >> 4, wx = rem & 15;
          const int iy  = n >> 3,  ix = n & 7;
          const size_t idx =
              (((size_t)bb << 14) + (wy * WS + iy) * WID + (wx * WS + ix)) * C_DIM + gc;
          v4f rv = *(const v4f*)&resid[idx];
          *(v4f*)&((float*)outp)[idx] = rv + val;
        } else {                                       // MODE 3
          const size_t idx = (size_t)gr * C_DIM + gc;
          v4f ov = *(const v4f*)&((float*)outp)[idx];
          *(v4f*)&((float*)outp)[idx] = ov + val;
        }
      }
      __syncthreads();
    }
  }
}

// ---------------------------------------------------------------------------
// Windowed attention: one block = (window, 2 heads); one wave = one head.
// tm-blocked: compute 16 query rows' S tiles, softmax them immediately,
// write P to LDS -> accumulator live range is 4x v8f (no spills).
// ---------------------------------------------------------------------------
__global__ __launch_bounds__(64) void attn_kernel(const half_t* __restrict__ qkv,
                                                  const float* __restrict__ bias8,
                                                  half_t* __restrict__ attn_out) {
  __shared__ half_t lq[2][NTOK * HD];
  __shared__ half_t lk[2][NTOK * HD];
  __shared__ half_t lv[2][NTOK * HD];
  __shared__ half_t lp[2][NTOK * 72];      // P, stride 72 (16B-aligned rows)
  __shared__ half_t lo[2][NTOK * HD];      // output stage

  const int win  = blockIdx.x >> 2;
  const int hp   = blockIdx.x & 3;
  const int wv   = threadIdx.x >> 5;
  const int lane = threadIdx.x & 31;
  const int head = hp * 2 + wv;

  {
    const size_t base = ((size_t)(win * 3) * NH + head) * (NTOK * HD);
    #pragma unroll
    for (int j = 0; j < 8; ++j) {
      *(v8h*)&lq[wv][lane * 64 + j * 8] = *(const v8h*)(qkv + base + lane * 64 + j * 8);
      *(v8h*)&lk[wv][lane * 64 + j * 8] =
          *(const v8h*)(qkv + base + (size_t)NH * NTOK * HD + lane * 64 + j * 8);
      *(v8h*)&lv[wv][lane * 64 + j * 8] =
          *(const v8h*)(qkv + base + (size_t)2 * NH * NTOK * HD + lane * 64 + j * 8);
    }
  }
  __syncthreads();

  const half_t* q  = lq[wv];
  const half_t* k  = lk[wv];
  const half_t* vp = lv[wv];
  half_t* p  = lp[wv];
  half_t* ob = lo[wv];
  const float* bias = bias8 + (size_t)head * NTOK * NTOK;

  const int ad0  = (lane < 16) ? 0 : 16;
  const int bd0  = (lane < 16) ? 0 : 16;
  const int mAdd = (lane < 16) ? 0 : 8;
  const int mcol = lane & 15;
  const float scale = 0.1767766952966369f;   // 32^-0.5

  // K fragments, reused across all tm blocks (4 x v16h)
  v16h kf[4];
  #pragma unroll
  for (int tn = 0; tn < 4; ++tn)
    kf[tn] = *(const v16h*)&k[(tn * 16 + mcol) * 32 + bd0];

  // ---- S = scale * q.k^T + bias, row-blocked softmax ----
  for (int tm = 0; tm < 4; ++tm) {
    const int arow = tm * 16 + mcol;
    v8h qlo = *(const v8h*)&q[arow * 32 + ad0];
    v8h qhi = *(const v8h*)&q[arow * 32 + ad0 + 8];
    v16h af;
    #pragma unroll
    for (int i = 0; i < 8; ++i) { af[i] = qlo[i]; af[i + 8] = qhi[i]; }

    v8f sa[4];
    #pragma unroll
    for (int tn = 0; tn < 4; ++tn) { v8f z = {}; sa[tn] = WMMA_F32_F16(af, kf[tn], z); }

    #pragma unroll
    for (int v = 0; v < 8; ++v) {
      const int n = tm * 16 + v + mAdd;
      float vals[4];
      #pragma unroll
      for (int tn = 0; tn < 4; ++tn)
        vals[tn] = sa[tn][v] * scale + bias[n * 64 + tn * 16 + mcol];
      float mx = fmaxf(fmaxf(vals[0], vals[1]), fmaxf(vals[2], vals[3]));
      #pragma unroll
      for (int off = 1; off < 16; off <<= 1) mx = fmaxf(mx, __shfl_xor(mx, off, 32));
      float sum = 0.0f;
      #pragma unroll
      for (int tn = 0; tn < 4; ++tn) { vals[tn] = __expf(vals[tn] - mx); sum += vals[tn]; }
      #pragma unroll
      for (int off = 1; off < 16; off <<= 1) sum += __shfl_xor(sum, off, 32);
      const float inv = 1.0f / sum;
      #pragma unroll
      for (int tn = 0; tn < 4; ++tn)
        p[n * 72 + tn * 16 + mcol] = (half_t)(vals[tn] * inv);
    }
  }
  __syncthreads();

  // ---- out = P (64x64) . v (64x32) ----
  v16h bfv[2][2];                                    // [kstep][tn], reused across tm
  #pragma unroll
  for (int ks = 0; ks < 2; ++ks)
    #pragma unroll
    for (int tn = 0; tn < 2; ++tn) {
      const int d  = tn * 16 + mcol;
      const int r0 = ks * 32 + bd0;
      v16h t;
      #pragma unroll
      for (int i = 0; i < 16; ++i) t[i] = vp[(r0 + i) * 32 + d];
      bfv[ks][tn] = t;
    }

  for (int tm = 0; tm < 4; ++tm) {
    v8f oa[2];
    { v8f z = {}; oa[0] = z; oa[1] = z; }
    #pragma unroll
    for (int ks = 0; ks < 2; ++ks) {
      const int arow = tm * 16 + mcol;
      const int c0 = ks * 32 + ad0;
      v8h plo = *(const v8h*)&p[arow * 72 + c0];
      v8h phi = *(const v8h*)&p[arow * 72 + c0 + 8];
      v16h af;
      #pragma unroll
      for (int i = 0; i < 8; ++i) { af[i] = plo[i]; af[i + 8] = phi[i]; }
      #pragma unroll
      for (int tn = 0; tn < 2; ++tn) oa[tn] = WMMA_F32_F16(af, bfv[ks][tn], oa[tn]);
    }
    #pragma unroll
    for (int tn = 0; tn < 2; ++tn)
      #pragma unroll
      for (int v = 0; v < 8; ++v)
        ob[(tm * 16 + v + mAdd) * 32 + tn * 16 + mcol] = (half_t)oa[tn][v];
  }
  __syncthreads();

  // ---- coalesced write: 8 x b128 per lane ----
  #pragma unroll
  for (int j = 0; j < 8; ++j) {
    const int s  = lane * 8 + j;                     // 256 v8h segments
    const int n  = s >> 2, d0 = (s & 3) * 8;
    *(v8h*)(attn_out + ((size_t)win * NTOK + n) * C_DIM + head * HD + d0) =
        *(const v8h*)&ob[n * 32 + d0];
  }
}

// ---------------------------------------------------------------------------
// Launch
// ---------------------------------------------------------------------------
extern "C" void kernel_launch(void* const* d_in, const int* in_sizes, int n_in,
                              void* d_out, int out_size, void* d_ws, size_t ws_size,
                              hipStream_t stream) {
  const float* x       = (const float*)d_in[0];
  const float* ln1_g   = (const float*)d_in[1];
  const float* ln1_b   = (const float*)d_in[2];
  const float* qkv_w   = (const float*)d_in[3];
  const float* qkv_b   = (const float*)d_in[4];
  const float* proj_w  = (const float*)d_in[5];
  const float* proj_b  = (const float*)d_in[6];
  const float* rel_tab = (const float*)d_in[7];
  const float* ln2_g   = (const float*)d_in[8];
  const float* ln2_b   = (const float*)d_in[9];
  const float* fc1_w   = (const float*)d_in[10];
  const float* fc1_b   = (const float*)d_in[11];
  const float* fc2_w   = (const float*)d_in[12];
  const float* fc2_b   = (const float*)d_in[13];
  const int*   rel_idx = (const int*)d_in[14];
  float* out = (float*)d_out;

  char* ws = (char*)d_ws;
  half_t* w_qkv16  = (half_t*)(ws + 0x0);            // 256*768
  half_t* w_proj16 = (half_t*)(ws + 0x60000);        // 256*256
  half_t* w_fc116  = (half_t*)(ws + 0x80000);        // 256*1024
  half_t* w_fc216  = (half_t*)(ws + 0x100000);       // 1024*256
  float*  bias8    = (float*)(ws + 0x140000);        // 8*64*64 f32 = 128KB
  half_t* yw16     = (half_t*)(ws + 0x180000);       // 131072*256 f16
  half_t* qkv16    = (half_t*)(ws + 0x4180000);      // max(qkv 192MB, h 256MB)
  half_t* attn16   = (half_t*)(ws + 0x14180000);     // 131072*256 f16
  half_t* h16      = qkv16;

  cvt_f16_kernel<<<(256 * 768 + 255) / 256, 256, 0, stream>>>(qkv_w, w_qkv16, 256 * 768);
  cvt_f16_kernel<<<(256 * 256 + 255) / 256, 256, 0, stream>>>(proj_w, w_proj16, 256 * 256);
  cvt_f16_kernel<<<(256 * 1024 + 255) / 256, 256, 0, stream>>>(fc1_w, w_fc116, 256 * 1024);
  cvt_f16_kernel<<<(1024 * 256 + 255) / 256, 256, 0, stream>>>(fc2_w, w_fc216, 1024 * 256);
  bias_expand_kernel<<<(NTOK * NTOK + 255) / 256, 256, 0, stream>>>(rel_tab, rel_idx, bias8);

  ln_kernel<<<MROWS, 256, 0, stream>>>(x, ln1_g, ln1_b, yw16, 1);

  gemm_wmma_kernel<0><<<dim3(768 / 64, MROWS / 128), 256, 0, stream>>>(
      yw16, w_qkv16, qkv_b, qkv16, nullptr, MROWS, 768, 256);

  attn_kernel<<<NWIN * 4, 64, 0, stream>>>(qkv16, bias8, attn16);

  gemm_wmma_kernel<1><<<dim3(256 / 64, MROWS / 128), 256, 0, stream>>>(
      attn16, w_proj16, proj_b, out, x, MROWS, 256, 256);

  ln_kernel<<<MROWS, 256, 0, stream>>>(out, ln2_g, ln2_b, yw16, 0);

  gemm_wmma_kernel<2><<<dim3(1024 / 64, MROWS / 128), 256, 0, stream>>>(
      yw16, w_fc116, fc1_b, h16, nullptr, MROWS, 1024, 256);

  gemm_wmma_kernel<3><<<dim3(256 / 64, MROWS / 128), 256, 0, stream>>>(
      h16, w_fc216, fc2_b, out, nullptr, MROWS, 256, 1024);
}
